// SpaceStateModel_35442070126826
// MI455X (gfx1250) — matrix operand order repaired and checked by reading, not compiled, not run on hardware
//
#include <hip/hip_runtime.h>
#include <math.h>

// ---------------------------------------------------------------------------
// SpaceStateModel (stacked LRU) for gfx1250 / MI455X.
// Heavy GEMMs (M=61440, K/N in {128,256}) run on v_wmma_f32_16x16x32_bf16,
// wave tile 16x64 (4 WMMAs share one A fragment). Sequential L=60 complex
// scans + small MLPs/LayerNorm/GLU run on VALU.
// ---------------------------------------------------------------------------

#define BATCH 1024
#define LVLS  60
#define NXF   9
#define NXSF  17
#define NH1   256
#define NH2   128
#define NYO   10
#define NYSO  8
#define UROWS (BATCH * LVLS)   // 61440

typedef __attribute__((ext_vector_type(16))) __bf16 v16bf;
typedef __attribute__((ext_vector_type(8)))  float  v8f;

// ---------------------------------------------------------------------------
// WMMA GEMM: out[m,n] = (BETA ? out : 0) + sum_k (±A[m,k]) * W[n,k]
//            (+ bias[n]) (* exp(gsc[n]))   ; NMASK guards n >= Nreal.
// Block: 128 threads (4 wave32). Wave w -> 16x(16*NT) tile at (m0+16w, n0).
// One A fragment feeds NT consecutive WMMAs (operand reuse in registers).
// ---------------------------------------------------------------------------
template<int NT, bool NEGA, bool BETA, bool BIAS, bool GSC, bool NMASK>
__global__ __launch_bounds__(128)
void wmma_gemm(const float* __restrict__ A, int lda,
               const float* __restrict__ W, int ldw,
               const float* __restrict__ bias,
               const float* __restrict__ gsc,
               float* __restrict__ out, int ldo,
               int K, int Nreal)
{
  const int lane = threadIdx.x & 31;
  const int wave = threadIdx.x >> 5;
  const int n0   = blockIdx.x * (16 * NT);
  const int m0   = blockIdx.y * 64 + wave * 16;

  const int r    = lane & 15;      // A row within tile / B,C column
  const int half = lane >> 4;      // lane half selects K sub-ranges
  const int kbA  = half * 16;      // A: K 0-15      vs 16-31
  const int kbB  = half * 16;      // B: K 0-15      vs 16-31

  const float* Arow = A + (size_t)(m0 + r) * lda;

  const float* Wrow[NT];
  bool nok[NT];
#pragma unroll
  for (int j = 0; j < NT; ++j) {
    const int n = n0 + 16 * j + r;
    nok[j]  = (!NMASK) || (n < Nreal);
    Wrow[j] = W + (size_t)n * ldw;
  }

  v8f acc[NT];
#pragma unroll
  for (int j = 0; j < NT; ++j) {
    if constexpr (BETA) {
#pragma unroll
      for (int i = 0; i < 8; ++i)
        acc[j][i] = nok[j]
            ? out[(size_t)(m0 + i + half * 8) * ldo + (n0 + 16 * j + r)]
            : 0.0f;
    } else {
#pragma unroll
      for (int i = 0; i < 8; ++i) acc[j][i] = 0.0f;
    }
  }

  for (int k0 = 0; k0 < K; k0 += 32) {
    // ---- A tile (16x32 bf16, contiguous K = 16*half + e), shared by NT WMMAs ----
    v16bf a;
    {
      const float* p = Arow + k0 + kbA;
      float4 f0 = *(const float4*)(p);
      float4 f1 = *(const float4*)(p + 4);
      float4 f2 = *(const float4*)(p + 8);
      float4 f3 = *(const float4*)(p + 12);
      float v[16] = {f0.x, f0.y, f0.z, f0.w, f1.x, f1.y, f1.z, f1.w,
                     f2.x, f2.y, f2.z, f2.w, f3.x, f3.y, f3.z, f3.w};
#pragma unroll
      for (int e = 0; e < 16; ++e)
        a[e] = (__bf16)(NEGA ? -v[e] : v[e]);     // native RTNE cvt
    }
    // ---- NT B tiles (32x16 bf16): column n = W row n, K contiguous ----
#pragma unroll
    for (int j = 0; j < NT; ++j) {
      v16bf b;
      if (nok[j]) {
        const float* p = Wrow[j] + k0 + kbB;
        float4 f0 = *(const float4*)(p);
        float4 f1 = *(const float4*)(p + 4);
        float4 f2 = *(const float4*)(p + 8);
        float4 f3 = *(const float4*)(p + 12);
        float v[16] = {f0.x, f0.y, f0.z, f0.w, f1.x, f1.y, f1.z, f1.w,
                       f2.x, f2.y, f2.z, f2.w, f3.x, f3.y, f3.z, f3.w};
#pragma unroll
        for (int e = 0; e < 16; ++e) b[e] = (__bf16)v[e];
      } else {
#pragma unroll
        for (int e = 0; e < 16; ++e) b[e] = (__bf16)0.0f;
      }
      acc[j] = __builtin_amdgcn_wmma_f32_16x16x32_bf16(
          false, a, false, b, (short)0, acc[j], false, false);
    }
  }

  // ---- epilogue ----
#pragma unroll
  for (int j = 0; j < NT; ++j) {
    const int nc = n0 + 16 * j + r;
    float bv = 0.0f, gv = 1.0f;
    if constexpr (BIAS) { if (nok[j]) bv = bias[nc]; }
    if constexpr (GSC)  { gv = expf(gsc[nc]); }
#pragma unroll
    for (int i = 0; i < 8; ++i) {
      float v = acc[j][i];
      if constexpr (BIAS) v += bv;
      if constexpr (GSC)  v *= gv;
      if (nok[j]) out[(size_t)(m0 + i + half * 8) * ldo + nc] = v;
    }
  }
}

template<int NT, bool NEGA, bool BETA, bool BIAS, bool GSC, bool NMASK>
static inline void launch_gemm(const float* A, int lda, const float* W, int ldw,
                               const float* bias, const float* gsc,
                               float* out, int ldo, int M, int K, int Nreal,
                               hipStream_t s)
{
  dim3 grid((Nreal + 16 * NT - 1) / (16 * NT), M / 64);
  wmma_gemm<NT, NEGA, BETA, BIAS, GSC, NMASK><<<grid, 128, 0, s>>>(
      A, lda, W, ldw, bias, gsc, out, ldo, K, Nreal);
}

// ---------------------------------------------------------------------------
// r1 = tanh(flip(inputs_main) @ Wi^T + bi)      (U x 128), K = 9
// ---------------------------------------------------------------------------
__global__ void k_r1(const float* __restrict__ xm, const float* __restrict__ Wi,
                     const float* __restrict__ bi, float* __restrict__ r1)
{
  int idx = blockIdx.x * blockDim.x + threadIdx.x;     // U*128
  int m = idx >> 7, j = idx & 127;
  int b = m / LVLS, l = m - b * LVLS;
  const float* x = xm + ((size_t)b * LVLS + (LVLS - 1 - l)) * NXF;
  const float* w = Wi + j * NXF;
  float acc = bi[j];
#pragma unroll
  for (int i = 0; i < NXF; ++i) acc += x[i] * w[i];
  r1[idx] = tanhf(acc);
}

// h0 = tanh(inputs_aux @ Ws1^T + bs1)            (B x 256), K = 17
__global__ void k_h0(const float* __restrict__ aux, const float* __restrict__ Ws1,
                     const float* __restrict__ bs1, float* __restrict__ h0)
{
  int idx = blockIdx.x * blockDim.x + threadIdx.x;     // B*256
  int b = idx >> 8, n = idx & 255;
  const float* a = aux + (size_t)b * NXSF;
  const float* w = Ws1 + (size_t)n * NXSF;
  float acc = bs1[n];
#pragma unroll
  for (int i = 0; i < NXSF; ++i) acc += a[i] * w[i];
  h0[idx] = tanhf(acc);
}

// Complex LRU scan over L: h = lam*h + Bu ; writes hs_{re,im}(B,L,N)
__global__ void k_scan(const float* __restrict__ bu_re, const float* __restrict__ bu_im,
                       const float* __restrict__ h0re,  // may be null (zero init)
                       const float* __restrict__ nu, const float* __restrict__ th,
                       float* __restrict__ hs_re, float* __restrict__ hs_im, int N)
{
  int idx = blockIdx.x * blockDim.x + threadIdx.x;     // B*N
  int b = idx / N, n = idx - b * N;
  float mag = expf(-expf(nu[n]));
  float ang = expf(th[n]);
  float lr = mag * cosf(ang), li = mag * sinf(ang);
  float hr = h0re ? h0re[(size_t)b * N + n] : 0.0f;
  float hi = 0.0f;
  size_t base = (size_t)b * LVLS * N + n;
  for (int l = 0; l < LVLS; ++l) {
    size_t o = base + (size_t)l * N;
    float nr = lr * hr - li * hi + bu_re[o];
    float ni = lr * hi + li * hr + bu_im[o];
    hr = nr; hi = ni;
    hs_re[o] = hr; hs_im[o] = hi;
  }
}

// Wave-per-row LayerNorm (N = 256 or 128). in==out allowed (same-wave RMW).
template<int N>
__global__ __launch_bounds__(128)
void k_ln(const float* __restrict__ in, float* __restrict__ outp,
          const float* __restrict__ s, const float* __restrict__ b)
{
  const int lane = threadIdx.x & 31;
  const int wave = threadIdx.x >> 5;
  const int row  = blockIdx.x * 4 + wave;
  constexpr int PER = N / 32;
  const float* x = in + (size_t)row * N + lane * PER;
  float v[PER]; float sum = 0.0f, sq = 0.0f;
#pragma unroll
  for (int i = 0; i < PER; ++i) { v[i] = x[i]; sum += v[i]; sq += v[i] * v[i]; }
#pragma unroll
  for (int o = 16; o > 0; o >>= 1) {
    sum += __shfl_xor(sum, o, 32);
    sq  += __shfl_xor(sq,  o, 32);
  }
  float mu  = sum / (float)N;
  float var = sq / (float)N - mu * mu;
  float inv = rsqrtf(var + 1e-5f);
  float* y = outp + (size_t)row * N + lane * PER;
#pragma unroll
  for (int i = 0; i < PER; ++i) {
    int c = lane * PER + i;
    y[i] = (v[i] - mu) * inv * s[c] + b[c];
  }
}

// GLU: z (U x 2H) -> s (U x H), optional flip along L
__global__ void k_glu(const float* __restrict__ z, float* __restrict__ s,
                      int H, int flip)
{
  int idx = blockIdx.x * blockDim.x + threadIdx.x;     // U*H
  int m = idx / H, j = idx - m * H;
  float a = z[(size_t)m * 2 * H + j];
  float g = z[(size_t)m * 2 * H + H + j];
  float val = a * (1.0f / (1.0f + expf(-g)));
  int mo = m;
  if (flip) { int b = m / LVLS, l = m - b * LVLS; mo = b * LVLS + (LVLS - 1 - l); }
  s[(size_t)mo * H + j] = val;
}

// out_sfc = y2[:, L-1] @ Wso^T + bso            (B x 8), K = 128
__global__ void k_sfc(const float* __restrict__ y2, const float* __restrict__ Wso,
                      const float* __restrict__ bso, float* __restrict__ outs)
{
  int idx = blockIdx.x * blockDim.x + threadIdx.x;     // B*8
  int b = idx >> 3, o = idx & 7;
  const float* row = y2 + ((size_t)b * LVLS + (LVLS - 1)) * NH2;
  const float* w = Wso + (size_t)o * NH2;
  float acc = bso[o];
  for (int k = 0; k < NH2; ++k) acc += row[k] * w[k];
  outs[idx] = acc;
}

// ---------------------------------------------------------------------------
extern "C" void kernel_launch(void* const* d_in, const int* in_sizes, int n_in,
                              void* d_out, int out_size, void* d_ws, size_t ws_size,
                              hipStream_t stream)
{
  (void)in_sizes; (void)n_in; (void)out_size; (void)ws_size;
  const float* inputs_main = (const float*)d_in[0];
  const float* inputs_aux  = (const float*)d_in[1];
  const float* Wi  = (const float*)d_in[2];
  const float* bi  = (const float*)d_in[3];
  const float* Ws1 = (const float*)d_in[4];
  const float* bs1 = (const float*)d_in[5];
  const float* nu1 = (const float*)d_in[6];
  const float* th1 = (const float*)d_in[7];
  const float* B1re = (const float*)d_in[8];
  const float* B1im = (const float*)d_in[9];
  const float* C1re = (const float*)d_in[10];
  const float* C1im = (const float*)d_in[11];
  const float* D1   = (const float*)d_in[12];
  const float* g1   = (const float*)d_in[13];
  const float* ln1s = (const float*)d_in[14];
  const float* ln1b = (const float*)d_in[15];
  const float* Wg1  = (const float*)d_in[16];
  const float* bg1  = (const float*)d_in[17];
  const float* nu2  = (const float*)d_in[18];
  const float* th2  = (const float*)d_in[19];
  const float* B2re = (const float*)d_in[20];
  const float* B2im = (const float*)d_in[21];
  const float* C2re = (const float*)d_in[22];
  const float* C2im = (const float*)d_in[23];
  const float* D2   = (const float*)d_in[24];
  const float* g2   = (const float*)d_in[25];
  const float* ln2s = (const float*)d_in[26];
  const float* ln2b = (const float*)d_in[27];
  const float* Wg2  = (const float*)d_in[28];
  const float* bg2  = (const float*)d_in[29];
  const float* Wl   = (const float*)d_in[30];
  const float* bl   = (const float*)d_in[31];
  const float* Wo   = (const float*)d_in[32];
  const float* bo   = (const float*)d_in[33];
  const float* Wso  = (const float*)d_in[34];
  const float* bso  = (const float*)d_in[35];

  const size_t U = (size_t)UROWS;
  float* ws = (float*)d_ws;
  // Aliased scratch pools (stages are strictly sequential on `stream`):
  float* r1 = ws;                 // U*128   (persists: feeds Bu1 + D1 terms)
  float* P1 = r1 + U * 128;       // U*256
  float* P2 = P1 + U * 256;       // U*256
  float* P3 = P2 + U * 256;       // U*256
  float* P4 = P3 + U * 256;       // U*256
  float* h0 = P4 + U * 256;       // B*256

  float* out_main = (float*)d_out;              // U*NYO
  float* out_sfc  = out_main + U * NYO;         // B*NYSO

  const int M = UROWS;

  // 1-2: input/aux tanh MLPs (tiny K -> VALU)
  k_r1<<<(U * 128) / 256, 256, 0, stream>>>(inputs_main, Wi, bi, r1);
  k_h0<<<(BATCH * NH1) / 256, 256, 0, stream>>>(inputs_aux, Ws1, bs1, h0);

  // 3-4: Bu1_{re,im} = r1 @ B1^T * exp(g1)   -> P1, P2
  launch_gemm<4, false, false, false, true, false>(r1, 128, B1re, 256, nullptr, g1, P1, 256, M, 128, 256, stream);
  launch_gemm<4, false, false, false, true, false>(r1, 128, B1im, 256, nullptr, g1, P2, 256, M, 128, 256, stream);

  // 5: scan1 -> hs1_{re,im} = P3, P4
  k_scan<<<(BATCH * NH1) / 256, 256, 0, stream>>>(P1, P2, h0, nu1, th1, P3, P4, NH1);

  // 6-8: y1 = hs_re@C1re^T - hs_im@C1im^T + r1@D1[:, :128]^T   -> P1
  launch_gemm<4, false, false, false, false, false>(P3, 256, C1re, 256, nullptr, nullptr, P1, 256, M, 256, 256, stream);
  launch_gemm<4, true,  true,  false, false, false>(P4, 256, C1im, 256, nullptr, nullptr, P1, 256, M, 256, 256, stream);
  launch_gemm<4, false, true,  false, false, false>(r1, 128, D1,   256, nullptr, nullptr, P1, 256, M, 128, 256, stream);

  // 9-11: GLU1 (LN in-place, linear, gate) + flip   -> s1 = P3 (lo half)
  k_ln<256><<<UROWS / 4, 128, 0, stream>>>(P1, P1, ln1s, ln1b);
  launch_gemm<4, false, false, true, false, false>(P1, 256, Wg1, 256, bg1, nullptr, P2, 256, M, 256, 256, stream);
  k_glu<<<(U * 128) / 256, 256, 0, stream>>>(P2, P3, 128, 1);

  // 12-13: Bu2 -> P4 halves
  float* Bu2re = P4; float* Bu2im = P4 + U * 128;
  launch_gemm<4, false, false, false, true, false>(P3, 128, B2re, 128, nullptr, g2, Bu2re, 128, M, 128, 128, stream);
  launch_gemm<4, false, false, false, true, false>(P3, 128, B2im, 128, nullptr, g2, Bu2im, 128, M, 128, 128, stream);

  // 14: scan2 (h0 = 0) -> hs2 in P1 halves
  float* hs2re = P1; float* hs2im = P1 + U * 128;
  k_scan<<<(BATCH * NH2) / 256, 256, 0, stream>>>(Bu2re, Bu2im, nullptr, nu2, th2, hs2re, hs2im, NH2);

  // 15-17: y2 -> P2 (lo half)
  float* y2 = P2;
  launch_gemm<4, false, false, false, false, false>(hs2re, 128, C2re, 128, nullptr, nullptr, y2, 128, M, 128, 128, stream);
  launch_gemm<4, true,  true,  false, false, false>(hs2im, 128, C2im, 128, nullptr, nullptr, y2, 128, M, 128, 128, stream);
  launch_gemm<4, false, true,  false, false, false>(P3,    128, D2,   128, nullptr, nullptr, y2, 128, M, 128, 128, stream);

  // 18-20: GLU2 (keep y2 intact for the surface head)  -> s2 = P3
  float* xn2 = P2 + U * 128;
  k_ln<128><<<UROWS / 4, 128, 0, stream>>>(y2, xn2, ln2s, ln2b);
  launch_gemm<4, false, false, true, false, false>(xn2, 128, Wg2, 128, bg2, nullptr, P4, 256, M, 128, 256, stream);
  k_glu<<<(U * 128) / 256, 256, 0, stream>>>(P4, P3, 128, 0);

  // 21: lat = s2 @ Wl^T + bl  -> P1 (lo)
  float* lat = P1;
  launch_gemm<4, false, false, true, false, false>(P3, 128, Wl, 128, bl, nullptr, lat, 128, M, 128, 128, stream);

  // 22: out = lat @ Wo^T + bo  (N=10, masked 16-wide tile) -> d_out
  launch_gemm<1, false, false, true, false, true>(lat, 128, Wo, 128, bo, nullptr, out_main, NYO, M, 128, NYO, stream);

  // 23: out_sfc
  k_sfc<<<(BATCH * NYSO) / 256, 256, 0, stream>>>(y2, Wso, bso, out_sfc);
}
